// MetaOmniBlock_65773129171579
// MI455X (gfx1250) — compile-verified, matrix operand order checked by reading, not run on hardware
//
#include <hip/hip_runtime.h>

typedef __attribute__((ext_vector_type(16))) __bf16 v16bf;
typedef __attribute__((ext_vector_type(8)))  float  v8f;
typedef __attribute__((ext_vector_type(4)))  __bf16 v4bf;

#define DEVINL __device__ __forceinline__

constexpr int   Bc   = 8;
constexpr int   Dc   = 384;
constexpr int   NTOK = 1024;        // 32*32 pixels
constexpr int   NHc  = 8;
constexpr int   DHc  = 48;
constexpr int   HIDc = 1536;
constexpr int   MTOT = Bc * NTOK;   // 8192 token rows
constexpr float EPSc = 1e-5f;

union Frag {
  uint4  u[2];
  v16bf  v;
  __bf16 h[16];
};

DEVINL v8f wmma_bf16(const Frag& a, const Frag& b, v8f c) {
  return __builtin_amdgcn_wmma_f32_16x16x32_bf16(false, a.v, false, b.v,
                                                 (short)0, c, false, false);
}

// ---------------------------------------------------------------- prep ----
__global__ __launch_bounds__(256) void wtrans_kernel(
    const float* __restrict__ w, __bf16* __restrict__ wt, int K, int N) {
  // w: [K][N] row-major (in,out) -> wt: [N][K] bf16 (out,in)
  int i = blockIdx.x * blockDim.x + threadIdx.x;
  if (i >= K * N) return;
  int n = i / K, k = i % K;
  wt[i] = (__bf16)w[(size_t)k * N + n];
}

__global__ __launch_bounds__(256) void cast_kernel(
    const float* __restrict__ src, __bf16* __restrict__ dst, int n) {
  int i = blockIdx.x * blockDim.x + threadIdx.x;
  if (i < n) dst[i] = (__bf16)src[i];
}

// ------------------------------------------------------- GroupNorm stats ----
__global__ __launch_bounds__(256) void gnstats_kernel(
    const float* __restrict__ src, float* __restrict__ stats) {
  constexpr int CNT = Dc * NTOK;  // 393216 per batch
  __shared__ float ss[256], sq[256];
  const int b = blockIdx.x, tid = threadIdx.x;
  const float* p = src + (size_t)b * CNT;
  float s = 0.f, q = 0.f;
  for (int i = tid; i < CNT; i += 256) {
    float v = p[i];
    s += v;
    q += v * v;
  }
  ss[tid] = s; sq[tid] = q;
  __syncthreads();
  for (int o = 128; o > 0; o >>= 1) {
    if (tid < o) { ss[tid] += ss[tid + o]; sq[tid] += sq[tid + o]; }
    __syncthreads();
  }
  if (tid == 0) {
    float m = ss[0] / CNT;
    float v = sq[0] / CNT - m * m;
    stats[b * 2 + 0] = m;
    stats[b * 2 + 1] = rsqrtf(v + EPSc);
  }
}

// ------------------------------------------------- transpose (b,c,n)->(b,n,c)
__global__ __launch_bounds__(256) void transpose_kernel(
    const float* __restrict__ x, float* __restrict__ xt) {
  __shared__ float tile[32][33];
  const int b = blockIdx.z, c0 = blockIdx.y * 32, n0 = blockIdx.x * 32;
  const int tx = threadIdx.x, ty = threadIdx.y;  // (32, 8)
#pragma unroll
  for (int j = 0; j < 4; ++j) {
    int c = ty + j * 8;
    tile[c][tx] = x[((size_t)b * Dc + c0 + c) * NTOK + n0 + tx];
  }
  __syncthreads();
#pragma unroll
  for (int j = 0; j < 4; ++j) {
    int n = ty + j * 8;
    xt[((size_t)b * NTOK + n0 + n) * Dc + c0 + tx] = tile[tx][n];
  }
}

// ------------------------------------------ GN1-affine + LayerNorm -> xn bf16
__global__ __launch_bounds__(256) void ln_kernel(
    const float* __restrict__ xt, const float* __restrict__ stats1,
    const float* __restrict__ g1, const float* __restrict__ b1,
    const float* __restrict__ lng, const float* __restrict__ lnb,
    __bf16* __restrict__ xn) {
  const int lane = threadIdx.x & 31, wid = threadIdx.x >> 5;
  const int tok = blockIdx.x * 8 + wid;  // 0..8191
  const int b = tok >> 10;
  const float m1 = stats1[b * 2], rs1 = stats1[b * 2 + 1];
  const float* row = xt + (size_t)tok * Dc;
  float hv[12];
  float s = 0.f, sq = 0.f;
#pragma unroll
  for (int i = 0; i < 3; ++i) {
    int c = (lane + 32 * i) * 4;
    float4 v = *(const float4*)&row[c];
    float t0 = (v.x - m1) * rs1 * g1[c + 0] + b1[c + 0];
    float t1 = (v.y - m1) * rs1 * g1[c + 1] + b1[c + 1];
    float t2 = (v.z - m1) * rs1 * g1[c + 2] + b1[c + 2];
    float t3 = (v.w - m1) * rs1 * g1[c + 3] + b1[c + 3];
    hv[i * 4 + 0] = t0; hv[i * 4 + 1] = t1;
    hv[i * 4 + 2] = t2; hv[i * 4 + 3] = t3;
    s += t0 + t1 + t2 + t3;
    sq += t0 * t0 + t1 * t1 + t2 * t2 + t3 * t3;
  }
#pragma unroll
  for (int o = 1; o < 32; o <<= 1) {
    s += __shfl_xor(s, o, 32);
    sq += __shfl_xor(sq, o, 32);
  }
  const float mean = s / Dc;
  const float rstd = rsqrtf(sq / Dc - mean * mean + EPSc);
#pragma unroll
  for (int i = 0; i < 3; ++i) {
    int c = (lane + 32 * i) * 4;
    v4bf o;
    o.x = (__bf16)((hv[i * 4 + 0] - mean) * rstd * lng[c + 0] + lnb[c + 0]);
    o.y = (__bf16)((hv[i * 4 + 1] - mean) * rstd * lng[c + 1] + lnb[c + 1]);
    o.z = (__bf16)((hv[i * 4 + 2] - mean) * rstd * lng[c + 2] + lnb[c + 2]);
    o.w = (__bf16)((hv[i * 4 + 3] - mean) * rstd * lng[c + 3] + lnb[c + 3]);
    *(v4bf*)&xn[(size_t)tok * Dc + c] = o;
  }
}

// ----------------------------------------------------------- tiled GEMM ----
// C[M,N] = A[M,K](bf16,row-major) * Wt[N,K]^T (bf16 [out][in])
// 128x64 macro-tile, 32x32 per wave (4 WMMA / K-step), double-buffered LDS.
// MODE 0: store bf16 (row*Ntot+col)              (q, k)
// MODE 1: store bf16 transposed to vT[b,h,d,tok] (v)
// MODE 2: f32 out = acc + resid  -> x2t          (wo + residual)
// MODE 3: bf16 out = GELU(acc + bias)            (fc1)
// MODE 4: f32 d_out[b,c,tok] = acc+bias+resid    (fc2 + residual + transpose)
constexpr int LDSS = 40;  // padded LDS row stride (bf16): 80B = 20 banks
template <int MODE>
__global__ __launch_bounds__(256) void gemm_wmma(
    const __bf16* __restrict__ A, const __bf16* __restrict__ Wt,
    const float* __restrict__ bias, const float* __restrict__ resid,
    void* __restrict__ outp, int K, int Ntot) {
  __shared__ __bf16 As[2][128 * LDSS];
  __shared__ __bf16 Bs[2][64 * LDSS];
  const int tid = threadIdx.x;
  const int lane = tid & 31, wid = tid >> 5;
  const int wm = wid & 3, wn = wid >> 2;         // 4 m-tiles x 2 n-tiles
  const int m0 = blockIdx.y * 128, n0 = blockIdx.x * 64;
  const int l15 = lane & 15, kb = (lane >> 4) * 8, half8 = (lane >> 4) * 8;

  auto stage = [&](int kk, int sel) {
#pragma unroll
    for (int i = 0; i < 2; ++i) {
      const int idx = tid + i * 256;              // 512 uint4 for A (128x32)
      const int row = idx >> 2, c8 = (idx & 3) * 8;
      *(uint4*)&As[sel][row * LDSS + c8] =
          *(const uint4*)&A[(size_t)(m0 + row) * K + kk + c8];
    }
    const int row = tid >> 2, c8 = (tid & 3) * 8;  // 256 uint4 for B (64x32)
    *(uint4*)&Bs[sel][row * LDSS + c8] =
        *(const uint4*)&Wt[(size_t)(n0 + row) * K + kk + c8];
  };

  v8f acc[2][2];
#pragma unroll
  for (int i = 0; i < 2; ++i)
#pragma unroll
    for (int j = 0; j < 2; ++j)
      acc[i][j] = v8f{0.f, 0.f, 0.f, 0.f, 0.f, 0.f, 0.f, 0.f};

  const int nk = K >> 5;
  stage(0, 0);
  for (int ks = 0; ks < nk; ++ks) {
    __syncthreads();
    if (ks + 1 < nk) stage((ks + 1) << 5, (ks + 1) & 1);
    const int sel = ks & 1;
    Frag a0, a1, b0, b1;
    const int mr = wm * 32 + l15, nr = wn * 32 + l15;
    a0.u[0] = *(const uint4*)&As[sel][mr * LDSS + kb];
    a0.u[1] = *(const uint4*)&As[sel][mr * LDSS + 16 + kb];
    a1.u[0] = *(const uint4*)&As[sel][(mr + 16) * LDSS + kb];
    a1.u[1] = *(const uint4*)&As[sel][(mr + 16) * LDSS + 16 + kb];
    b0.u[0] = *(const uint4*)&Bs[sel][nr * LDSS + kb];
    b0.u[1] = *(const uint4*)&Bs[sel][nr * LDSS + 16 + kb];
    b1.u[0] = *(const uint4*)&Bs[sel][(nr + 16) * LDSS + kb];
    b1.u[1] = *(const uint4*)&Bs[sel][(nr + 16) * LDSS + 16 + kb];
    acc[0][0] = wmma_bf16(a0, b0, acc[0][0]);
    acc[0][1] = wmma_bf16(a0, b1, acc[0][1]);
    acc[1][0] = wmma_bf16(a1, b0, acc[1][0]);
    acc[1][1] = wmma_bf16(a1, b1, acc[1][1]);
  }

#pragma unroll
  for (int mi = 0; mi < 2; ++mi) {
#pragma unroll
    for (int ni = 0; ni < 2; ++ni) {
      const int col = n0 + wn * 32 + ni * 16 + l15;
      const int rowb = m0 + wm * 32 + mi * 16 + half8;
#pragma unroll
      for (int r = 0; r < 8; ++r) {
        const int row = rowb + r;
        float v = acc[mi][ni][r];
        if (MODE == 0) {
          ((__bf16*)outp)[(size_t)row * Ntot + col] = (__bf16)v;
        } else if (MODE == 1) {
          const int bb = row >> 10, tok = row & 1023;
          const int hh = col / DHc, dd = col % DHc;
          ((__bf16*)outp)[((size_t)((bb * NHc + hh) * DHc + dd)) * NTOK + tok] =
              (__bf16)v;
        } else if (MODE == 2) {
          v += resid[(size_t)row * Ntot + col];
          ((float*)outp)[(size_t)row * Ntot + col] = v;
        } else if (MODE == 3) {
          v += bias[col];
          v = 0.5f * v * (1.0f + erff(v * 0.70710678118654752f));
          ((__bf16*)outp)[(size_t)row * Ntot + col] = (__bf16)v;
        } else {  // MODE 4
          v += bias[col] + resid[(size_t)row * Dc + col];
          const int bb = row >> 10, tok = row & 1023;
          ((float*)outp)[((size_t)(bb * Dc + col)) * NTOK + tok] = v;
        }
      }
    }
  }
}

// ----------------------------------------------- depthwise 3x3 gate conv ----
__global__ __launch_bounds__(256) void gate_conv_kernel(
    const __bf16* __restrict__ xn, const float* __restrict__ gw,
    const float* __restrict__ gb, float* __restrict__ gate) {
  int i = blockIdx.x * blockDim.x + threadIdx.x;
  if (i >= Bc * NTOK * Dc) return;
  const int c = i % Dc;
  const int n = (i / Dc) & (NTOK - 1);
  const int b = i / (Dc * NTOK);
  const int y = n >> 5, x = n & 31;
  float acc = gb[c];
#pragma unroll
  for (int dy = -1; dy <= 1; ++dy) {
    const int yy = y + dy;
    if (yy < 0 || yy >= 32) continue;
#pragma unroll
    for (int dx = -1; dx <= 1; ++dx) {
      const int xx = x + dx;
      if (xx < 0 || xx >= 32) continue;
      const float w = gw[c * 9 + (dy + 1) * 3 + (dx + 1)];
      acc += w * (float)xn[((size_t)(b * NTOK + (yy << 5) + xx)) * Dc + c];
    }
  }
  gate[i] = acc;
}

// ---------------------------------------- flash attention, S^T formulation --
// one wave: (b, h, 16-query block), 32 keys per iteration.
// S^T = K*Q^T; the two S^T sub-tiles land exactly in the low/high halves of
// the P^T B-fragment, so Y^T += V^T * P^T runs a full K=32 with no padding
// and no cross-lane movement.  The Q fragment's zero half (d=48..63) masks
// the K fragment's high half, so K's d=48..63 can be garbage loads (no
// per-iteration zero-register rebuilds).
__global__ __launch_bounds__(256) void attention_kernel(
    const __bf16* __restrict__ q, const __bf16* __restrict__ k,
    const __bf16* __restrict__ vT, const float* __restrict__ gate,
    __bf16* __restrict__ yg) {
  const int lane = threadIdx.x & 31, wid = threadIdx.x >> 5;
  const int gw = blockIdx.x * 8 + wid;      // 0..4095
  const int qb = gw & 63;
  const int h = (gw >> 6) & 7;
  const int b = gw >> 9;
  const int half = lane >> 4, l15 = lane & 15, kb8 = half * 8;
  const int c0 = h * DHc;
  const int nq = qb * 16 + l15;
  const uint4 zero4 = {0u, 0u, 0u, 0u};
  const float scale = 0.144337567297406441f;  // 1/sqrt(48)

  // Q fragments (B-operand): lane = query column, d 8*half..+7 and 16+8*half..+7 per lane.
  // q1's high half is ZERO: it masks the d=48..63 K-dim for both operands.
  const __bf16* qrow = q + (size_t)(b * NTOK + nq) * Dc + c0;
  Frag q0, q1;
  q0.u[0] = *(const uint4*)(qrow + kb8);
  q0.u[1] = *(const uint4*)(qrow + 16 + kb8);
  q1.u[0] = *(const uint4*)(qrow + 32 + kb8);  // d 32..47
  q1.u[1] = zero4;                             // d 48..63 == 0 (the mask)

  v8f yt[3];
#pragma unroll
  for (int t = 0; t < 3; ++t)
    yt[t] = v8f{0.f, 0.f, 0.f, 0.f, 0.f, 0.f, 0.f, 0.f};
  float rmax = -1e30f, rsum = 0.f;
  Frag pf;

  const __bf16* vbase = vT + ((size_t)(b * NHc + h) * DHc) * NTOK;

  for (int kblk = 0; kblk < 32; ++kblk) {
    const int nk = kblk * 32 + l15;
    const __bf16* krow0 = k + (size_t)(b * NTOK + nk) * Dc + c0;
    const __bf16* krow1 = krow0 + (size_t)16 * Dc;
    Frag k0a, k0b, k1a, k1b;
    k0a.u[0] = *(const uint4*)(krow0 + kb8);
    k0a.u[1] = *(const uint4*)(krow0 + 16 + kb8);
    k0b.u[0] = *(const uint4*)(krow0 + 32 + kb8);
    k0b.u[1] = *(const uint4*)(krow0 + 48 + kb8);  // garbage, masked by q1
    k1a.u[0] = *(const uint4*)(krow1 + kb8);
    k1a.u[1] = *(const uint4*)(krow1 + 16 + kb8);
    k1b.u[0] = *(const uint4*)(krow1 + 32 + kb8);
    k1b.u[1] = *(const uint4*)(krow1 + 48 + kb8);  // garbage, masked by q1
    if (kblk + 1 < 32) {
      __builtin_prefetch(krow0 + (size_t)32 * Dc, 0, 3);
      __builtin_prefetch(vbase + (size_t)l15 * NTOK + (kblk + 1) * 32, 0, 3);
    }

    v8f st0 = {0.f, 0.f, 0.f, 0.f, 0.f, 0.f, 0.f, 0.f};
    v8f st1 = {0.f, 0.f, 0.f, 0.f, 0.f, 0.f, 0.f, 0.f};
    // interleave the two independent accumulation chains
    st0 = wmma_bf16(k0a, q0, st0);   // keys 0..15, d 0..31
    st1 = wmma_bf16(k1a, q0, st1);   // keys 16..31, d 0..31
    st0 = wmma_bf16(k0b, q1, st0);   // keys 0..15, d 32..47 (+masked)
    st1 = wmma_bf16(k1b, q1, st1);   // keys 16..31, d 32..47 (+masked)

    float bm = -1e30f;
#pragma unroll
    for (int r = 0; r < 8; ++r) {
      st0[r] *= scale;
      st1[r] *= scale;
      bm = fmaxf(bm, fmaxf(st0[r], st1[r]));
    }
    bm = fmaxf(bm, __shfl_xor(bm, 16, 32));   // combine the two key halves
    const float nmax = fmaxf(rmax, bm);
    const float fac = __expf(rmax - nmax);
    float ps = 0.f;
#pragma unroll
    for (int r = 0; r < 8; ++r) {
      const float p0 = __expf(st0[r] - nmax);
      const float p1 = __expf(st1[r] - nmax);
      ps += p0 + p1;
      pf.h[r] = (__bf16)p0;       // k = kb8 + r        (sub-tile 0)
      pf.h[8 + r] = (__bf16)p1;   // k = 16 + kb8 + r   (sub-tile 1)
    }
    ps += __shfl_xor(ps, 16, 32);
    rsum = rsum * fac + ps;
    rmax = nmax;

#pragma unroll
    for (int t = 0; t < 3; ++t) {
#pragma unroll
      for (int r = 0; r < 8; ++r) yt[t][r] *= fac;
    }
#pragma unroll
    for (int t = 0; t < 3; ++t) {
      Frag vf;
      const __bf16* vp = vbase + (size_t)(t * 16 + l15) * NTOK + kblk * 32;
      vf.u[0] = *(const uint4*)(vp + kb8);
      vf.u[1] = *(const uint4*)(vp + 16 + kb8);
      yt[t] = wmma_bf16(vf, pf, yt[t]);       // Y^T += V^T * P^T (full K=32)
    }
  }

  const float inv = 1.f / rsum;
  const float* grow = gate + (size_t)(b * NTOK + nq) * Dc + c0;
  __bf16* orow = yg + (size_t)(b * NTOK + nq) * Dc + c0;
#pragma unroll
  for (int t = 0; t < 3; ++t) {
#pragma unroll
    for (int r = 0; r < 8; ++r) {
      const int d = t * 16 + half * 8 + r;
      const float g = grow[d];
      const float sg = 1.f / (1.f + __expf(-g));
      orow[d] = (__bf16)(yt[t][r] * inv * sg);
    }
  }
}

// --------------------------------------------------- GN2 affine -> h2 bf16 --
__global__ __launch_bounds__(256) void gn2_apply_kernel(
    const float* __restrict__ x2t, const float* __restrict__ stats2,
    const float* __restrict__ g2, const float* __restrict__ b2,
    __bf16* __restrict__ h2) {
  int i = blockIdx.x * blockDim.x + threadIdx.x;
  if (i >= Bc * NTOK * Dc) return;
  const int c = i % Dc;
  const int b = i / (Dc * NTOK);
  const float m = stats2[b * 2], rs = stats2[b * 2 + 1];
  h2[i] = (__bf16)((x2t[i] - m) * rs * g2[c] + b2[c]);
}

// ----------------------------------------------------------------- launch ---
extern "C" void kernel_launch(void* const* d_in, const int* in_sizes, int n_in,
                              void* d_out, int out_size, void* d_ws,
                              size_t ws_size, hipStream_t stream) {
  const float* x      = (const float*)d_in[0];
  const float* ln_g   = (const float*)d_in[1];
  const float* ln_b   = (const float*)d_in[2];
  const float* wq     = (const float*)d_in[3];
  const float* wk     = (const float*)d_in[4];
  const float* wv     = (const float*)d_in[5];
  const float* wo     = (const float*)d_in[6];
  const float* gate_w = (const float*)d_in[7];
  const float* gate_b = (const float*)d_in[8];
  const float* gn1_g  = (const float*)d_in[9];
  const float* gn1_b  = (const float*)d_in[10];
  const float* gn2_g  = (const float*)d_in[11];
  const float* gn2_b  = (const float*)d_in[12];
  const float* fc1_w  = (const float*)d_in[13];
  const float* fc1_b  = (const float*)d_in[14];
  const float* fc2_w  = (const float*)d_in[15];
  const float* fc2_b  = (const float*)d_in[16];

  char* ws = (char*)d_ws;
  const size_t SZ_WT   = (size_t)Dc * Dc * 2;        // 294912
  const size_t SZ_FCW  = (size_t)HIDc * Dc * 2;      // 1179648
  const size_t SZ_F32  = (size_t)MTOT * Dc * 4;      // 12582912
  const size_t SZ_BF   = (size_t)MTOT * Dc * 2;      // 6291456

  size_t off = 0;
  __bf16* wqT   = (__bf16*)(ws + off); off += SZ_WT;
  __bf16* wkT   = (__bf16*)(ws + off); off += SZ_WT;
  __bf16* wvT   = (__bf16*)(ws + off); off += SZ_WT;
  __bf16* woT   = (__bf16*)(ws + off); off += SZ_WT;
  __bf16* fc1wB = (__bf16*)(ws + off); off += SZ_FCW;
  __bf16* fc2wB = (__bf16*)(ws + off); off += SZ_FCW;
  float*  st1   = (float*)(ws + off);  off += 256;
  float*  st2   = (float*)(ws + off);  off += 256;
  float*  xt    = (float*)(ws + off);  size_t o_xt = off; off += SZ_F32;
  __bf16* xn    = (__bf16*)(ws + off); off += SZ_BF;
  __bf16* qb    = (__bf16*)(ws + off); off += SZ_BF;
  __bf16* kb    = (__bf16*)(ws + off); off += SZ_BF;
  __bf16* vT    = (__bf16*)(ws + off); off += SZ_BF;
  float*  gate  = (float*)(ws + off);  off += SZ_F32;
  __bf16* yg    = (__bf16*)(ws + off); off += SZ_BF;
  float*  x2t   = (float*)(ws + off);  off += SZ_F32;
  __bf16* h2    = (__bf16*)(ws + off); off += SZ_BF;
  // h3 aliases the dead xt/xn/q region (they are unused after the WO GEMM)
  __bf16* h3    = (__bf16*)(ws + o_xt);               // 8192*1536*2 bytes

  const int EW = (Bc * NTOK * Dc + 255) / 256;        // 12288 blocks

  // 1. weight prep (cast + transpose to [out][in] bf16)
  wtrans_kernel<<<(Dc * Dc + 255) / 256, 256, 0, stream>>>(wq, wqT, Dc, Dc);
  wtrans_kernel<<<(Dc * Dc + 255) / 256, 256, 0, stream>>>(wk, wkT, Dc, Dc);
  wtrans_kernel<<<(Dc * Dc + 255) / 256, 256, 0, stream>>>(wv, wvT, Dc, Dc);
  wtrans_kernel<<<(Dc * Dc + 255) / 256, 256, 0, stream>>>(wo, woT, Dc, Dc);
  cast_kernel<<<(HIDc * Dc + 255) / 256, 256, 0, stream>>>(fc1_w, fc1wB, HIDc * Dc);
  cast_kernel<<<(HIDc * Dc + 255) / 256, 256, 0, stream>>>(fc2_w, fc2wB, HIDc * Dc);

  // 2. GN1 stats, transpose x -> token-major, fused GN1-affine + LN
  gnstats_kernel<<<Bc, 256, 0, stream>>>(x, st1);
  transpose_kernel<<<dim3(32, 12, Bc), dim3(32, 8), 0, stream>>>(x, xt);
  ln_kernel<<<MTOT / 8, 256, 0, stream>>>(xt, st1, gn1_g, gn1_b, ln_g, ln_b, xn);

  // 3. QKV projections (V stored pre-transposed per head)
  gemm_wmma<0><<<dim3(Dc / 64, MTOT / 128), 256, 0, stream>>>(xn, wqT, nullptr, nullptr, qb, Dc, Dc);
  gemm_wmma<0><<<dim3(Dc / 64, MTOT / 128), 256, 0, stream>>>(xn, wkT, nullptr, nullptr, kb, Dc, Dc);
  gemm_wmma<1><<<dim3(Dc / 64, MTOT / 128), 256, 0, stream>>>(xn, wvT, nullptr, nullptr, vT, Dc, Dc);

  // 4. depthwise gate conv on xn
  gate_conv_kernel<<<EW, 256, 0, stream>>>(xn, gate_w, gate_b, gate);

  // 5. flash attention + sigmoid gate
  attention_kernel<<<(Bc * NHc * 64) / 8, 256, 0, stream>>>(qb, kb, vT, gate, yg);

  // 6. WO projection + residual -> x2t
  gemm_wmma<2><<<dim3(Dc / 64, MTOT / 128), 256, 0, stream>>>(yg, woT, nullptr, xt, x2t, Dc, Dc);

  // 7. GN2 + MLP (fc1+GELU, fc2+bias+residual, store transposed to d_out)
  gnstats_kernel<<<Bc, 256, 0, stream>>>(x2t, st2);
  gn2_apply_kernel<<<EW, 256, 0, stream>>>(x2t, st2, gn2_g, gn2_b, h2);
  gemm_wmma<3><<<dim3(HIDc / 64, MTOT / 128), 256, 0, stream>>>(h2, fc1wB, fc1_b, nullptr, h3, Dc, HIDc);
  gemm_wmma<4><<<dim3(Dc / 64, MTOT / 128), 256, 0, stream>>>(h3, fc2wB, fc2_b, x2t, d_out, HIDc, Dc);
}
